// PoseSolver_6897717477669
// MI455X (gfx1250) — matrix operand order repaired and checked by reading, not compiled, not run on hardware
//
#include <hip/hip_runtime.h>
#include <hip/hip_bf16.h>
#include <math.h>

// ---------------------------------------------------------------------------
// Types for CDNA5 WMMA (wave32): v_wmma_f32_16x16x32_f16
// ---------------------------------------------------------------------------
typedef _Float16 vhalf16 __attribute__((ext_vector_type(16)));
typedef _Float16 vhalf8  __attribute__((ext_vector_type(8)));
typedef float    vfloat8 __attribute__((ext_vector_type(8)));

#define NRAW   2048
#define KSEL   256
#define NTOK   512        // 2*KSEL
#define DM     256
#define DG     64
#define NHEAD  4
#define DH     64
#define FDIM   768
#define SCALE  0.125f     // DH^-0.5

__device__ __forceinline__ float gelu_f(float x) {
    return 0.5f * x * (1.0f + erff(x * 0.70710678118654752f));
}

// ---------------------------------------------------------------------------
// Generic NT WMMA GEMM: C[M,N] = act(A[M,K] * Bt[N,K]^T + bias) (+ residual)
// One wave32 per 16x16 tile; 8 waves / 256-thread workgroup.
// A,Bt are f16; accumulate f32 in WMMA; optional f16 and/or f32 output.
// ---------------------------------------------------------------------------
template<int ACT, bool HAS_BIAS, bool HAS_RES, bool OUT16, bool OUT32>
__global__ __launch_bounds__(256) void gemm_nt(
    const _Float16* __restrict__ A,  int lda,
    const _Float16* __restrict__ Bt, int ldb,
    const float*    __restrict__ bias,
    const float*    __restrict__ res, int ldr,
    _Float16* __restrict__ C16, float* __restrict__ C32, int ldc,
    int M, int N, int K)
{
    int wave = (blockIdx.x * 256 + threadIdx.x) >> 5;
    int lane = threadIdx.x & 31;
    int ntn  = N >> 4;
    int ntiles = (M >> 4) * ntn;
    if (wave >= ntiles) return;
    int tm = (wave / ntn) << 4;
    int tn = (wave % ntn) << 4;

    int arow = tm + (lane & 15);
    int ak   = (lane >> 4) << 3;   // 0 or 8  (K offset of lo chunk)
    int bcol = tn + (lane & 15);
    int bk   = (lane >> 4) << 3;   // 0 or 8  (K offset of lo chunk)

    vfloat8 acc = {0.f,0.f,0.f,0.f,0.f,0.f,0.f,0.f};
    for (int k0 = 0; k0 < K; k0 += 32) {
        const vhalf8* pa0 = (const vhalf8*)(A + (size_t)arow * lda + k0 + ak);
        const vhalf8* pa1 = (const vhalf8*)(A + (size_t)arow * lda + k0 + ak + 16);
        const vhalf8* pb  = (const vhalf8*)(Bt + (size_t)bcol * ldb + k0 + bk);
        vhalf8 alo = pa0[0], ahi = pa1[0];
        vhalf8 blo = pb[0],  bhi = pb[2];
        if (k0 + 32 < K) {
            __builtin_prefetch((const void*)(A + (size_t)arow * lda + k0 + 32 + ak), 0, 1);
            __builtin_prefetch((const void*)(Bt + (size_t)bcol * ldb + k0 + 32 + bk), 0, 1);
        }
        vhalf16 af, bf;
        #pragma unroll
        for (int i = 0; i < 8; ++i) {
            af[i] = alo[i]; af[8 + i] = ahi[i];
            bf[i] = blo[i]; bf[8 + i] = bhi[i];
        }
        acc = __builtin_amdgcn_wmma_f32_16x16x32_f16(false, af, false, bf,
                                                     (short)0, acc, false, false);
    }

    int mbase = tm + ((lane >> 4) << 3);
    int col   = tn + (lane & 15);
    #pragma unroll
    for (int r = 0; r < 8; ++r) {
        float v = acc[r];
        if (HAS_BIAS) v += bias[col];
        if (ACT == 1) v = gelu_f(v);
        int m = mbase + r;
        if (HAS_RES) v += res[(size_t)m * ldr + col];
        if (OUT16) C16[(size_t)m * ldc + col] = (_Float16)v;
        if (OUT32) C32[(size_t)m * ldc + col] = v;
    }
}

// ---------------------------------------------------------------------------
// Fused geometry MLP: per 128 (i,j) pairs: [dist, rel/dist] -> 64 (VALU gelu,
// into LDS) -> 64 (WMMA from LDS). Avoids a 32MB intermediate.
// ---------------------------------------------------------------------------
__global__ __launch_bounds__(256) void geom_fused(
    const float* __restrict__ pts,   // 512 x 3
    const float* __restrict__ w0,    // (4,64)
    const float* __restrict__ b0,    // 64
    const _Float16* __restrict__ w1T,// (64,64) transposed f16
    const float* __restrict__ b1,    // 64
    _Float16* __restrict__ geom)     // (512*512) x 64
{
    __shared__ _Float16 sh[128 * 64];
    int base = blockIdx.x * 128;
    int tid  = threadIdx.x;
    {
        int r = tid >> 1, g0 = (tid & 1) * 32;
        int p = base + r;
        int i = p >> 9, j = p & 511;
        float rx = pts[i*3+0] - pts[j*3+0];
        float ry = pts[i*3+1] - pts[j*3+1];
        float rz = pts[i*3+2] - pts[j*3+2];
        float dist = sqrtf(rx*rx + ry*ry + rz*rz);
        float inv  = 1.0f / (dist + 1e-6f);
        float i0 = dist, i1 = rx*inv, i2 = ry*inv, i3 = rz*inv;
        #pragma unroll 8
        for (int g = g0; g < g0 + 32; ++g) {
            float a = b0[g] + i0*w0[g] + i1*w0[64+g] + i2*w0[128+g] + i3*w0[192+g];
            sh[r*64 + g] = (_Float16)gelu_f(a);
        }
    }
    __syncthreads();

    int wave = tid >> 5, lane = tid & 31;
    #pragma unroll
    for (int it = 0; it < 4; ++it) {
        int tile = wave + it * 8;          // 8x4 = 32 tiles of 16x16
        int tm = (tile >> 2) << 4, tn = (tile & 3) << 4;
        int arow = tm + (lane & 15);
        int ak   = (lane >> 4) << 3;
        int bcol = tn + (lane & 15);
        int bk   = (lane >> 4) << 3;
        vfloat8 acc = {0.f,0.f,0.f,0.f,0.f,0.f,0.f,0.f};
        #pragma unroll
        for (int k0 = 0; k0 < 64; k0 += 32) {
            const vhalf8* pa0 = (const vhalf8*)(sh + arow*64 + k0 + ak);
            const vhalf8* pa1 = (const vhalf8*)(sh + arow*64 + k0 + ak + 16);
            const vhalf8* pb  = (const vhalf8*)(w1T + bcol*64 + k0 + bk);
            vhalf8 alo = pa0[0], ahi = pa1[0];
            vhalf8 blo = pb[0],  bhi = pb[2];
            vhalf16 af, bf;
            #pragma unroll
            for (int i = 0; i < 8; ++i) {
                af[i] = alo[i]; af[8+i] = ahi[i];
                bf[i] = blo[i]; bf[8+i] = bhi[i];
            }
            acc = __builtin_amdgcn_wmma_f32_16x16x32_f16(false, af, false, bf,
                                                         (short)0, acc, false, false);
        }
        int mbase = tm + ((lane >> 4) << 3);
        int col   = tn + (lane & 15);
        #pragma unroll
        for (int r = 0; r < 8; ++r) {
            float v = acc[r] + b1[col];
            geom[(size_t)(base + mbase + r) * 64 + col] = (_Float16)v;
        }
    }
}

// ---------------------------------------------------------------------------
// Weight prep: f32 (din,dout) -> f16 transposed (dout,din); plain convert.
// ---------------------------------------------------------------------------
__global__ void wconv_t(const float* __restrict__ W, _Float16* __restrict__ WT,
                        int din, int dout) {
    int idx = blockIdx.x * 256 + threadIdx.x;
    if (idx >= din * dout) return;
    int i = idx / dout, o = idx % dout;
    WT[(size_t)o * din + i] = (_Float16)W[idx];
}
__global__ void wconv(const float* __restrict__ W, _Float16* __restrict__ Wc, int n) {
    int idx = blockIdx.x * 256 + threadIdx.x;
    if (idx < n) Wc[idx] = (_Float16)W[idx];
}
__global__ void transpose_h(const _Float16* __restrict__ src, _Float16* __restrict__ dst,
                            int rows, int cols) {
    int idx = blockIdx.x * 256 + threadIdx.x;
    if (idx >= rows * cols) return;
    int r = idx / cols, c = idx % cols;
    dst[(size_t)c * rows + r] = src[idx];
}

// ---------------------------------------------------------------------------
// Top-K=256 by confidence (descending, ties -> lower index). O(N^2) ranking.
// Also gathers points/conf and precomputes log(conf+1e-6).
// ---------------------------------------------------------------------------
__global__ __launch_bounds__(256) void topk_select(
    const float* __restrict__ pa, const float* __restrict__ ca,
    const float* __restrict__ pb, const float* __restrict__ cb,
    int* __restrict__ sel, float* __restrict__ conf_sel,
    float* __restrict__ lc, float* __restrict__ pts_sel)
{
    int view = blockIdx.x;
    const float* conf = view ? cb : ca;
    const float* pts  = view ? pb : pa;
    __shared__ float sc[NRAW];
    for (int i = threadIdx.x; i < NRAW; i += 256) sc[i] = conf[i];
    __syncthreads();
    for (int s = 0; s < NRAW / 256; ++s) {
        int i = threadIdx.x + 256 * s;
        float ci = sc[i];
        int rank = 0;
        for (int m = 0; m < NRAW; ++m) {
            float cm = sc[m];
            rank += (cm > ci) || (cm == ci && m < i);
        }
        if (rank < KSEL) {
            int slot = view * KSEL + rank;
            sel[slot] = i;
            conf_sel[slot] = ci;
            lc[slot] = logf(ci + 1e-6f);
            pts_sel[slot*3+0] = pts[i*3+0];
            pts_sel[slot*3+1] = pts[i*3+1];
            pts_sel[slot*3+2] = pts[i*3+2];
        }
    }
}

__global__ void gather_feats(const float* __restrict__ fa, const float* __restrict__ fb,
                             const int* __restrict__ sel, _Float16* __restrict__ out) {
    int idx = blockIdx.x * 256 + threadIdx.x;
    if (idx >= NTOK * FDIM) return;
    int t = idx / FDIM, f = idx % FDIM;
    const float* src = (t < KSEL) ? fa : fb;
    out[idx] = (_Float16)src[(size_t)sel[t] * FDIM + f];
}

// tiny K<32 first layers: VALU + gelu
__global__ void point_l1(const float* __restrict__ p, const float* __restrict__ w,
                         const float* __restrict__ b, _Float16* __restrict__ out) {
    int idx = blockIdx.x * 256 + threadIdx.x;
    if (idx >= NTOK * 64) return;
    int t = idx >> 6, g = idx & 63;
    float a = b[g] + p[t*3]*w[g] + p[t*3+1]*w[64+g] + p[t*3+2]*w[128+g];
    out[idx] = (_Float16)gelu_f(a);
}
__global__ void conf_l1(const float* __restrict__ c, const float* __restrict__ w,
                        const float* __restrict__ b, _Float16* __restrict__ out) {
    int idx = blockIdx.x * 256 + threadIdx.x;
    if (idx >= NTOK * 32) return;
    int t = idx >> 5, g = idx & 31;
    out[idx] = (_Float16)gelu_f(b[g] + c[t] * w[g]);
}

// gate * x + view_emb, then LayerNorm -> f32 token stream
__global__ __launch_bounds__(256) void embed_post(
    const float* __restrict__ xt, const float* __restrict__ conf_sel,
    const float* __restrict__ alpha, const float* __restrict__ beta,
    const float* __restrict__ vemb, const float* __restrict__ g,
    const float* __restrict__ bb, float* __restrict__ x)
{
    int t = blockIdx.x, d = threadIdx.x;
    float cf = conf_sel[t];
    float gate = 1.0f / (1.0f + expf(-(alpha[0]*cf + beta[0])));
    float y = xt[t*DM + d] * gate + vemb[(t < KSEL ? 0 : 1) * DM + d];
    __shared__ float red[256];
    red[d] = y; __syncthreads();
    for (int s = 128; s > 0; s >>= 1) { if (d < s) red[d] += red[d+s]; __syncthreads(); }
    float mean = red[0] * (1.0f/256.0f); __syncthreads();
    float dv = y - mean;
    red[d] = dv*dv; __syncthreads();
    for (int s = 128; s > 0; s >>= 1) { if (d < s) red[d] += red[d+s]; __syncthreads(); }
    float var = red[0] * (1.0f/256.0f);
    x[t*DM + d] = dv * rsqrtf(var + 1e-5f) * g[d] + bb[d];
}

__global__ __launch_bounds__(256) void ln_f16(
    const float* __restrict__ x, const float* __restrict__ g,
    const float* __restrict__ bb, _Float16* __restrict__ out)
{
    int t = blockIdx.x, d = threadIdx.x;
    float y = x[t*DM + d];
    __shared__ float red[256];
    red[d] = y; __syncthreads();
    for (int s = 128; s > 0; s >>= 1) { if (d < s) red[d] += red[d+s]; __syncthreads(); }
    float mean = red[0] * (1.0f/256.0f); __syncthreads();
    float dv = y - mean;
    red[d] = dv*dv; __syncthreads();
    for (int s = 128; s > 0; s >>= 1) { if (d < s) red[d] += red[d+s]; __syncthreads(); }
    float var = red[0] * (1.0f/256.0f);
    out[t*DM + d] = (_Float16)(dv * rsqrtf(var + 1e-5f) * g[d] + bb[d]);
}

// w[i, h*64+g] = scale*qg + gb[g,h];  qb[i,h] = scale * q_i,h . kg_bias_h
__global__ void wprep(const float* __restrict__ qg, const float* __restrict__ gbw,
                      float* __restrict__ wb) {
    int idx = blockIdx.x * 256 + threadIdx.x;
    if (idx >= NTOK * DM) return;
    int hg = idx & 255; int g = hg & 63; int h = hg >> 6;
    wb[idx] = SCALE * qg[idx] + gbw[g*NHEAD + h];
}
__global__ void qbprep(const _Float16* __restrict__ q16, const float* __restrict__ kgb,
                       float* __restrict__ qb) {
    int idx = blockIdx.x * 256 + threadIdx.x;
    if (idx >= NTOK * NHEAD) return;
    int t = idx >> 2, h = idx & 3;
    float s = 0.f;
    for (int d = 0; d < DH; ++d) s += (float)q16[t*DM + h*DH + d] * kgb[h*DH + d];
    qb[idx] = SCALE * s;
}

// logits = scale*qk + geom.w + qb + gamma*(lc_i+lc_j); softmax over j -> f16
__global__ __launch_bounds__(256) void attn_softmax(
    const float* __restrict__ qk, const _Float16* __restrict__ geom,
    const float* __restrict__ wb, const float* __restrict__ qb,
    const float* __restrict__ lc, const float* __restrict__ gamma,
    _Float16* __restrict__ attn)
{
    int i = blockIdx.x, tid = threadIdx.x;
    __shared__ float wsh[256];
    __shared__ float red[256];
    wsh[tid] = wb[i*DM + tid];
    __syncthreads();
    float gm = gamma[0], lci = lc[i];
    float lg[2][NHEAD];
    #pragma unroll
    for (int s = 0; s < 2; ++s) {
        int j = tid + 256 * s;
        const _Float16* gr = geom + ((size_t)i * NTOK + j) * DG;
        float dot[NHEAD] = {0.f, 0.f, 0.f, 0.f};
        for (int g = 0; g < DG; ++g) {
            float gv = (float)gr[g];
            #pragma unroll
            for (int h = 0; h < NHEAD; ++h) dot[h] += gv * wsh[h*DG + g];
        }
        float lcj = gm * (lci + lc[j]);
        #pragma unroll
        for (int h = 0; h < NHEAD; ++h)
            lg[s][h] = SCALE * qk[((size_t)h * NTOK + i) * NTOK + j]
                     + dot[h] + qb[i*NHEAD + h] + lcj;
    }
    #pragma unroll
    for (int h = 0; h < NHEAD; ++h) {
        red[tid] = fmaxf(lg[0][h], lg[1][h]); __syncthreads();
        for (int s = 128; s > 0; s >>= 1) {
            if (tid < s) red[tid] = fmaxf(red[tid], red[tid+s]); __syncthreads();
        }
        float mx = red[0]; __syncthreads();
        float e0 = expf(lg[0][h] - mx), e1 = expf(lg[1][h] - mx);
        red[tid] = e0 + e1; __syncthreads();
        for (int s = 128; s > 0; s >>= 1) {
            if (tid < s) red[tid] += red[tid+s]; __syncthreads();
        }
        float inv = 1.0f / red[0]; __syncthreads();
        _Float16* row = attn + ((size_t)h * NTOK + i) * NTOK;
        row[tid]       = (_Float16)(e0 * inv);
        row[tid + 256] = (_Float16)(e1 * inv);
    }
}

// attention pooling per view
__global__ __launch_bounds__(256) void pool_kernel(
    const float* __restrict__ x, const float* __restrict__ pq, float* __restrict__ out)
{
    int view = blockIdx.x, tid = threadIdx.x;
    const float* xv = x + (size_t)view * KSEL * DM;
    float s = 0.f;
    for (int d = 0; d < DM; ++d) s += xv[tid*DM + d] * pq[d];
    __shared__ float red[256];
    __shared__ float ev[256];
    red[tid] = s; __syncthreads();
    for (int t = 128; t > 0; t >>= 1) {
        if (tid < t) red[tid] = fmaxf(red[tid], red[tid+t]); __syncthreads();
    }
    float mx = red[0]; __syncthreads();
    float e = expf(s - mx);
    ev[tid] = e; red[tid] = e; __syncthreads();
    for (int t = 128; t > 0; t >>= 1) {
        if (tid < t) red[tid] += red[tid+t]; __syncthreads();
    }
    float inv = 1.0f / red[0]; __syncthreads();
    float acc = 0.f;
    for (int t = 0; t < KSEL; ++t) acc += ev[t] * xv[t*DM + tid];
    out[view*DM + tid] = acc * inv;
}

// fused head: concat -> m1(gelu) -> m2 -> rot6d/trans, writes d_out[0..11]
__global__ __launch_bounds__(256) void head_kernel(
    const float* __restrict__ pooled,
    const float* __restrict__ m1w, const float* __restrict__ m1b,
    const float* __restrict__ m2w, const float* __restrict__ m2b,
    const float* __restrict__ rw,  const float* __restrict__ rb,
    const float* __restrict__ tw,  const float* __restrict__ tb,
    float* __restrict__ out)
{
    __shared__ float z[4*DM];
    __shared__ float y1[DM];
    __shared__ float y2[DM];
    __shared__ float r6[6];
    __shared__ float t3[3];
    int tid = threadIdx.x;
    float a = pooled[tid], b = pooled[DM + tid];
    z[tid] = a; z[DM + tid] = b; z[2*DM + tid] = a - b; z[3*DM + tid] = a * b;
    __syncthreads();
    float s = m1b[tid];
    for (int i = 0; i < 4*DM; ++i) s += z[i] * m1w[i*DM + tid];
    y1[tid] = gelu_f(s); __syncthreads();
    s = m2b[tid];
    for (int i = 0; i < DM; ++i) s += y1[i] * m2w[i*DM + tid];
    y2[tid] = s; __syncthreads();
    if (tid < 6)  { float v = rb[tid]; for (int i = 0; i < DM; ++i) v += y2[i]*rw[i*6+tid]; r6[tid] = v; }
    if (tid >= 8 && tid < 11) { int c = tid-8; float v = tb[c]; for (int i = 0; i < DM; ++i) v += y2[i]*tw[i*3+c]; t3[c] = v; }
    __syncthreads();
    if (tid == 0) {
        float a1[3] = {r6[0], r6[1], r6[2]}, a2[3] = {r6[3], r6[4], r6[5]};
        float n1 = fmaxf(sqrtf(a1[0]*a1[0]+a1[1]*a1[1]+a1[2]*a1[2]), 1e-12f);
        float b1[3] = {a1[0]/n1, a1[1]/n1, a1[2]/n1};
        float dp = b1[0]*a2[0] + b1[1]*a2[1] + b1[2]*a2[2];
        float p[3] = {a2[0]-dp*b1[0], a2[1]-dp*b1[1], a2[2]-dp*b1[2]};
        float n2 = fmaxf(sqrtf(p[0]*p[0]+p[1]*p[1]+p[2]*p[2]), 1e-12f);
        float b2[3] = {p[0]/n2, p[1]/n2, p[2]/n2};
        float b3[3] = {b1[1]*b2[2]-b1[2]*b2[1], b1[2]*b2[0]-b1[0]*b2[2], b1[0]*b2[1]-b1[1]*b2[0]};
        out[0]=b1[0]; out[1]=b1[1]; out[2]=b1[2];
        out[3]=b2[0]; out[4]=b2[1]; out[5]=b2[2];
        out[6]=b3[0]; out[7]=b3[1]; out[8]=b3[2];
        float tn = fmaxf(sqrtf(t3[0]*t3[0]+t3[1]*t3[1]+t3[2]*t3[2]), 1e-12f);
        out[9]=t3[0]/tn; out[10]=t3[1]/tn; out[11]=t3[2]/tn;
    }
}

__global__ void copy_tokens(const float* __restrict__ x, float* __restrict__ out) {
    int idx = blockIdx.x * 256 + threadIdx.x;
    if (idx < NTOK * DM) out[12 + idx] = x[idx];
}

// ---------------------------------------------------------------------------
// Host orchestration
// ---------------------------------------------------------------------------
extern "C" void kernel_launch(void* const* d_in, const int* in_sizes, int n_in,
                              void* d_out, int out_size, void* d_ws, size_t ws_size,
                              hipStream_t stream)
{
    (void)in_sizes; (void)n_in; (void)out_size; (void)ws_size;
    auto P = [&](int i) { return (const float*)d_in[i]; };

    // param leaf indices (jax tree order: keys sorted alphabetically)
    // blocks[b] base = 6 + 20*b:
    //  +0 f1.b +1 f1.w +2 f2.b +3 f2.w +4 gamma +5 gb.w +6 k.b +7 k.w +8 kg.b
    //  +9 kg.w +10 n1.b +11 n1.g +12 n2.b +13 n2.g +14 out.b +15 out.w
    //  +16 q.b +17 q.w +18 v.b +19 v.w
    const int EA = 86;  // embed.alpha
    const int EB = 87, C0B = 88, C0W = 89, C1B = 90, C1W = 91;
    const int F0B = 92, F0W = 93, F1B = 94, F1W = 95, LNB = 96, LNG = 97;
    const int P0B = 98, P0W = 99, P1B = 100, P1W = 101, PRB = 102, PRW = 103;
    const int VEMB = 104, G0B = 105, G0W = 106, G1B = 107, G1W = 108;
    const int M1B = 109, M1W = 110, M2B = 111, M2W = 112;
    const int RB = 113, RW = 114, TB = 115, TW = 116, PQ = 117;

    char* wsb = (char*)d_ws; size_t off = 0;
    auto alloc = [&](size_t bytes) -> char* {
        off = (off + 255) & ~(size_t)255;
        char* p = wsb + off; off += bytes; return p;
    };

    int*      sel      = (int*)alloc(NTOK * 4);
    float*    pts_sel  = (float*)alloc(NTOK * 3 * 4);
    float*    conf_sel = (float*)alloc(NTOK * 4);
    float*    lc       = (float*)alloc(NTOK * 4);
    float*    x        = (float*)alloc(NTOK * DM * 4);
    float*    xtmp     = (float*)alloc(NTOK * DM * 4);
    _Float16* hbuf     = (_Float16*)alloc(NTOK * DM * 2);
    _Float16* q16      = (_Float16*)alloc(NTOK * DM * 2);
    _Float16* k16      = (_Float16*)alloc(NTOK * DM * 2);
    _Float16* v16      = (_Float16*)alloc(NTOK * DM * 2);
    _Float16* vT16     = (_Float16*)alloc(NTOK * DM * 2);
    _Float16* o16      = (_Float16*)alloc(NTOK * DM * 2);
    _Float16* f1buf    = (_Float16*)alloc(NTOK * 1024 * 2);
    _Float16* cat16    = (_Float16*)alloc(NTOK * 224 * 2);
    _Float16* pe1      = (_Float16*)alloc(NTOK * 64 * 2);
    _Float16* ce1      = (_Float16*)alloc(NTOK * 32 * 2);
    _Float16* fe1      = (_Float16*)alloc(NTOK * 128 * 2);
    _Float16* feats16  = (_Float16*)alloc(NTOK * FDIM * 2);
    float*    qg       = (float*)alloc(NTOK * DM * 4);
    float*    wbuf     = (float*)alloc(NTOK * DM * 4);
    float*    qb       = (float*)alloc(NTOK * NHEAD * 4);
    float*    pooled   = (float*)alloc(2 * DM * 4);
    float*    qk       = (float*)alloc((size_t)NHEAD * NTOK * NTOK * 4);
    _Float16* attn16   = (_Float16*)alloc((size_t)NHEAD * NTOK * NTOK * 2);
    _Float16* fw0T     = (_Float16*)alloc((size_t)FDIM * 128 * 2);
    _Float16* fw1T     = (_Float16*)alloc(128 * 128 * 2);
    _Float16* pw1T     = (_Float16*)alloc(64 * 64 * 2);
    _Float16* cw1T     = (_Float16*)alloc(32 * 32 * 2);
    _Float16* prT      = (_Float16*)alloc(224 * 256 * 2);
    _Float16* g1T      = (_Float16*)alloc(64 * 64 * 2);
    _Float16 *wqT[4], *wkT[4], *wvT[4], *woT[4], *wf1T[4], *wf2T[4], *wkg[4];
    for (int b = 0; b < 4; ++b) {
        wqT[b]  = (_Float16*)alloc(DM * DM * 2);
        wkT[b]  = (_Float16*)alloc(DM * DM * 2);
        wvT[b]  = (_Float16*)alloc(DM * DM * 2);
        woT[b]  = (_Float16*)alloc(DM * DM * 2);
        wf1T[b] = (_Float16*)alloc((size_t)DM * 1024 * 2);
        wf2T[b] = (_Float16*)alloc((size_t)1024 * DM * 2);
        wkg[b]  = (_Float16*)alloc(DG * DM * 2);
    }
    _Float16* geom16 = (_Float16*)alloc((size_t)NTOK * NTOK * DG * 2);

    auto convT = [&](int idx, _Float16* dst, int din, int dout) {
        int n = din * dout;
        wconv_t<<<(n + 255) / 256, 256, 0, stream>>>(P(idx), dst, din, dout);
    };
    auto tiles = [&](int M, int N) { return ((M >> 4) * (N >> 4) + 7) / 8; };

    // ---- weight conversion (every call; deterministic) ----
    convT(F0W, fw0T, FDIM, 128);
    convT(F1W, fw1T, 128, 128);
    convT(P1W, pw1T, 64, 64);
    convT(C1W, cw1T, 32, 32);
    convT(PRW, prT, 224, 256);
    convT(G1W, g1T, 64, 64);
    for (int b = 0; b < 4; ++b) {
        int B = 6 + 20 * b;
        convT(B + 17, wqT[b], DM, DM);
        convT(B + 7,  wkT[b], DM, DM);
        convT(B + 19, wvT[b], DM, DM);
        convT(B + 15, woT[b], DM, DM);
        convT(B + 1,  wf1T[b], DM, 1024);
        convT(B + 3,  wf2T[b], 1024, DM);
        wconv<<<(DG * DM + 255) / 256, 256, 0, stream>>>(P(B + 9), wkg[b], DG * DM);
    }

    // ---- embed: top-k select, gather, MLPs ----
    topk_select<<<2, 256, 0, stream>>>(P(0), P(2), P(3), P(5), sel, conf_sel, lc, pts_sel);
    gather_feats<<<(NTOK * FDIM + 255) / 256, 256, 0, stream>>>(P(1), P(4), sel, feats16);
    point_l1<<<(NTOK * 64 + 255) / 256, 256, 0, stream>>>(pts_sel, P(P0W), P(P0B), pe1);
    conf_l1<<<(NTOK * 32 + 255) / 256, 256, 0, stream>>>(conf_sel, P(C0W), P(C0B), ce1);

    gemm_nt<1, true, false, true, false><<<tiles(NTOK, 128), 256, 0, stream>>>(
        feats16, FDIM, fw0T, FDIM, P(F0B), nullptr, 0, fe1, nullptr, 128, NTOK, 128, FDIM);
    gemm_nt<0, true, false, true, false><<<tiles(NTOK, 128), 256, 0, stream>>>(
        fe1, 128, fw1T, 128, P(F1B), nullptr, 0, cat16 + 64, nullptr, 224, NTOK, 128, 128);
    gemm_nt<0, true, false, true, false><<<tiles(NTOK, 64), 256, 0, stream>>>(
        pe1, 64, pw1T, 64, P(P1B), nullptr, 0, cat16, nullptr, 224, NTOK, 64, 64);
    gemm_nt<0, true, false, true, false><<<tiles(NTOK, 32), 256, 0, stream>>>(
        ce1, 32, cw1T, 32, P(C1B), nullptr, 0, cat16 + 192, nullptr, 224, NTOK, 32, 32);
    gemm_nt<0, true, false, false, true><<<tiles(NTOK, DM), 256, 0, stream>>>(
        cat16, 224, prT, 224, P(PRB), nullptr, 0, nullptr, xtmp, DM, NTOK, DM, 224);
    embed_post<<<NTOK, 256, 0, stream>>>(xtmp, conf_sel, P(EA), P(EB), P(VEMB),
                                         P(LNG), P(LNB), x);

    // ---- fused geometry MLP: (512*512, 4) -> 64 -> 64, f16 ----
    geom_fused<<<(NTOK * NTOK) / 128, 256, 0, stream>>>(
        pts_sel, P(G0W), P(G0B), g1T, P(G1B), geom16);

    // ---- transformer blocks ----
    for (int b = 0; b < 4; ++b) {
        int B = 6 + 20 * b;
        ln_f16<<<NTOK, 256, 0, stream>>>(x, P(B + 11), P(B + 10), hbuf);
        gemm_nt<0, true, false, true, false><<<tiles(NTOK, DM), 256, 0, stream>>>(
            hbuf, DM, wqT[b], DM, P(B + 16), nullptr, 0, q16, nullptr, DM, NTOK, DM, DM);
        gemm_nt<0, true, false, true, false><<<tiles(NTOK, DM), 256, 0, stream>>>(
            hbuf, DM, wkT[b], DM, P(B + 6), nullptr, 0, k16, nullptr, DM, NTOK, DM, DM);
        gemm_nt<0, true, false, true, false><<<tiles(NTOK, DM), 256, 0, stream>>>(
            hbuf, DM, wvT[b], DM, P(B + 18), nullptr, 0, v16, nullptr, DM, NTOK, DM, DM);
        transpose_h<<<(NTOK * DM + 255) / 256, 256, 0, stream>>>(v16, vT16, NTOK, DM);

        // qg[i,h,g] = q_h . Wkg_h (folds 512x512x64x256 gk tensor away)
        for (int h = 0; h < NHEAD; ++h)
            gemm_nt<0, false, false, false, true><<<tiles(NTOK, DG), 256, 0, stream>>>(
                q16 + h * DH, DM, wkg[b] + h * DH, DM, nullptr, nullptr, 0,
                nullptr, qg + h * DH, DM, NTOK, DG, DH);
        wprep<<<(NTOK * DM + 255) / 256, 256, 0, stream>>>(qg, P(B + 5), wbuf);
        qbprep<<<(NTOK * NHEAD + 255) / 256, 256, 0, stream>>>(q16, P(B + 8), qb);

        // qk[h,i,j] = q_h . k_h^T
        for (int h = 0; h < NHEAD; ++h)
            gemm_nt<0, false, false, false, true><<<tiles(NTOK, NTOK), 256, 0, stream>>>(
                q16 + h * DH, DM, k16 + h * DH, DM, nullptr, nullptr, 0,
                nullptr, qk + (size_t)h * NTOK * NTOK, NTOK, NTOK, NTOK, DH);

        attn_softmax<<<NTOK, 256, 0, stream>>>(qk, geom16, wbuf, qb, lc, P(B + 4), attn16);

        // o_h = attn_h @ v_h  (v transposed -> NT form)
        for (int h = 0; h < NHEAD; ++h)
            gemm_nt<0, false, false, true, false><<<tiles(NTOK, DH), 256, 0, stream>>>(
                attn16 + (size_t)h * NTOK * NTOK, NTOK, vT16 + (size_t)h * DH * NTOK, NTOK,
                nullptr, nullptr, 0, o16 + h * DH, nullptr, DM, NTOK, DH, NTOK);

        gemm_nt<0, true, true, false, true><<<tiles(NTOK, DM), 256, 0, stream>>>(
            o16, DM, woT[b], DM, P(B + 14), x, DM, nullptr, x, DM, NTOK, DM, DM);

        ln_f16<<<NTOK, 256, 0, stream>>>(x, P(B + 13), P(B + 12), hbuf);
        gemm_nt<1, true, false, true, false><<<tiles(NTOK, 1024), 256, 0, stream>>>(
            hbuf, DM, wf1T[b], DM, P(B + 0), nullptr, 0, f1buf, nullptr, 1024, NTOK, 1024, DM);
        gemm_nt<0, true, true, false, true><<<tiles(NTOK, DM), 256, 0, stream>>>(
            f1buf, 1024, wf2T[b], 1024, P(B + 2), x, DM, nullptr, x, DM, NTOK, DM, 1024);
    }

    // ---- outputs: R(9), tr(3), ta(256x256), tb(256x256) ----
    float* out = (float*)d_out;
    copy_tokens<<<(NTOK * DM + 255) / 256, 256, 0, stream>>>(x, out);
    pool_kernel<<<2, 256, 0, stream>>>(x, P(PQ), pooled);
    head_kernel<<<1, 256, 0, stream>>>(pooled, P(M1W), P(M1B), P(M2W), P(M2B),
                                       P(RW), P(RB), P(TW), P(TB), out);
}
